// SparseAutoencoder_80753975099585
// MI455X (gfx1250) — matrix operand order repaired and checked by reading, not compiled, not run on hardware
//
#include <hip/hip_runtime.h>
#include <hip/hip_bf16.h>

#define D_MODEL 768
#define D_SAE   24576
#define K_TOP   32
#define BATCH   4096

#define BM 64
#define BN 256
#define BK 32
#define LDA 40  // padded k-stride (u16) for A tile in LDS
#define LDB 40  // padded k-stride (u16) for transposed B tile in LDS

typedef __attribute__((ext_vector_type(16))) __bf16 v16bf;
typedef __attribute__((ext_vector_type(8)))  float  v8f;
typedef __attribute__((ext_vector_type(8)))  unsigned short u16x8;
typedef __attribute__((ext_vector_type(4)))  unsigned short u16x4;

union FragBF {
  v16bf bf;
  u16x8 h[2];
};

__device__ __forceinline__ unsigned short f2bf(float f) {
  unsigned int u = __float_as_uint(f);
  unsigned int r = (u + 0x7FFFu + ((u >> 16) & 1u)) >> 16;  // round-to-nearest-even
  return (unsigned short)r;
}

// ---------------------------------------------------------------- convert
__global__ void convert_f32_to_bf16(const float* __restrict__ src,
                                    unsigned short* __restrict__ dst, int n) {
  int i = blockIdx.x * blockDim.x + threadIdx.x;
  int stride = gridDim.x * blockDim.x;
  for (; i < n; i += stride) dst[i] = f2bf(src[i]);
}

// ---------------------------------------------------------------- encode GEMM (WMMA bf16)
// C[BATCH][D_SAE] = A(bf16)[BATCH][D_MODEL] @ B(bf16)[D_MODEL][D_SAE] + b_enc
__global__ __launch_bounds__(256)
void encode_gemm_wmma(const unsigned short* __restrict__ A,
                      const unsigned short* __restrict__ B,
                      const float* __restrict__ b_enc,
                      float* __restrict__ C) {
  __shared__ __align__(16) unsigned short ldsA[2][BM * LDA];  // 2 x 5 KB, row-major [m][k]
  __shared__ __align__(16) unsigned short ldsB[2][BN * LDB];  // 2 x 20 KB, transposed [n][k]

  const int t    = threadIdx.x;
  const int lane = t & 31;
  const int w    = t >> 5;     // 0..7
  const int wm   = w >> 2;     // 0..1 -> 32 rows each
  const int wn   = w & 3;      // 0..3 -> 64 cols each
  const int mBlock = blockIdx.y * BM;
  const int nBlock = blockIdx.x * BN;

  const int hl  = lane >> 4;   // K-half selector per ISA 16-bit A/B layout
  const int l15 = lane & 15;   // row (A) / col (B,C)

  v8f acc[2][4] = {};

  // A staging: thread -> (row = t>>2, 8-elem k-chunk = (t&3)*8)
  const int aRow = t >> 2, aChunk = (t & 3) * 8;
  // B staging: thread -> 8(K) x 4(N) micro-tile, transposed in registers
  const int bKG = (t >> 6) * 8;   // k-group base: 0,8,16,24
  const int bN0 = (t & 63) * 4;   // n base: 0..252

  u16x8 ra;
  u16x4 rb[8];

  // ---- prologue: fetch tile 0 and stage into buffer 0
  ra = *reinterpret_cast<const u16x8*>(
      &A[(size_t)(mBlock + aRow) * D_MODEL + aChunk]);
#pragma unroll
  for (int j = 0; j < 8; ++j)
    rb[j] = *reinterpret_cast<const u16x4*>(
        &B[(size_t)(bKG + j) * D_SAE + nBlock + bN0]);
  *reinterpret_cast<u16x8*>(&ldsA[0][aRow * LDA + aChunk]) = ra;
#pragma unroll
  for (int i = 0; i < 4; ++i) {
    u16x8 o;
#pragma unroll
    for (int j = 0; j < 8; ++j) o[j] = rb[j][i];
    *reinterpret_cast<u16x8*>(&ldsB[0][(bN0 + i) * LDB + bKG]) = o;  // b128 store
  }
  __syncthreads();

  const int NIT = D_MODEL / BK;  // 24
  for (int it = 0; it < NIT; ++it) {
    const int cur = it & 1, nxt = cur ^ 1;

    // ---- prefetch next tile into the other buffer (overlaps WMMAs below)
    if (it + 1 < NIT) {
      const int kb = (it + 1) * BK;
      ra = *reinterpret_cast<const u16x8*>(
          &A[(size_t)(mBlock + aRow) * D_MODEL + kb + aChunk]);
#pragma unroll
      for (int j = 0; j < 8; ++j)
        rb[j] = *reinterpret_cast<const u16x4*>(
            &B[(size_t)(kb + bKG + j) * D_SAE + nBlock + bN0]);
      *reinterpret_cast<u16x8*>(&ldsA[nxt][aRow * LDA + aChunk]) = ra;
#pragma unroll
      for (int i = 0; i < 4; ++i) {
        u16x8 o;
#pragma unroll
        for (int j = 0; j < 8; ++j) o[j] = rb[j][i];
        *reinterpret_cast<u16x8*>(&ldsB[nxt][(bN0 + i) * LDB + bKG]) = o;
      }
    }

    // ---- fragments from current buffer
    FragBF a[2], b[4];
#pragma unroll
    for (int mi = 0; mi < 2; ++mi) {
      int row = wm * 32 + mi * 16 + l15;
      // lanes 0-15: K 0..7 & 16..23 ; lanes 16-31: K 8..15 & 24..31
      a[mi].h[0] = *reinterpret_cast<const u16x8*>(&ldsA[cur][row * LDA + hl * 8]);
      a[mi].h[1] = *reinterpret_cast<const u16x8*>(&ldsA[cur][row * LDA + 16 + hl * 8]);
    }
#pragma unroll
    for (int ni = 0; ni < 4; ++ni) {
      int col = wn * 64 + ni * 16 + l15;
      // lanes 0-15: K 0..7 & 16..23 ; lanes 16-31: K 8..15 & 24..31
      b[ni].h[0] = *reinterpret_cast<const u16x8*>(&ldsB[cur][col * LDB + hl * 8]);
      b[ni].h[1] = *reinterpret_cast<const u16x8*>(&ldsB[cur][col * LDB + 16 + hl * 8]);
    }
#pragma unroll
    for (int mi = 0; mi < 2; ++mi)
#pragma unroll
      for (int ni = 0; ni < 4; ++ni)
        acc[mi][ni] = __builtin_amdgcn_wmma_f32_16x16x32_bf16(
            false, a[mi].bf, false, b[ni].bf, (short)0, acc[mi][ni], false, false);

    // single barrier per K-step: separates this iter's reads of buf[cur]
    // from next iter's writes to buf[cur]
    __syncthreads();
  }

  // epilogue: + b_enc, write pre_acts (C/D layout: lane&15=N, lane>>4 selects M 0-7 / 8-15)
#pragma unroll
  for (int ni = 0; ni < 4; ++ni) {
    int col = nBlock + wn * 64 + ni * 16 + l15;
    float bias = b_enc[col];
#pragma unroll
    for (int mi = 0; mi < 2; ++mi) {
      int mBase = mBlock + wm * 32 + mi * 16 + hl * 8;
#pragma unroll
      for (int j = 0; j < 8; ++j)
        C[(size_t)(mBase + j) * D_SAE + col] = acc[mi][ni][j] + bias;
    }
  }
}

// ---------------------------------------------------------------- top-k + scatter + sparse decode
__global__ __launch_bounds__(256)
void topk_decode(float* __restrict__ acts,        // in: pre_acts row, out: sparse acts row
                 const float* __restrict__ x,
                 const float* __restrict__ W_dec,
                 const float* __restrict__ b_dec,
                 float* __restrict__ xrec,
                 float* __restrict__ idx_out,
                 float* __restrict__ part_rec,
                 float* __restrict__ part_spa) {
  __shared__ float cv[256 * 32];
  __shared__ int   ci[256 * 32];
  __shared__ float wvs[8]; __shared__ int wis[8]; __shared__ int wls[8];
  __shared__ float top_v[K_TOP]; __shared__ int top_i[K_TOP];
  __shared__ float rv[K_TOP];
  __shared__ int   sel;
  __shared__ float red[8];

  const int t   = threadIdx.x;
  const int row = blockIdx.x;
  float* __restrict__ pre = acts + (size_t)row * D_SAE;

  float* mycv = cv + t * 32;
  int*   myci = ci + t * 32;

  // phase 1: per-thread top-32 over 96 strided elements (replace-min)
#pragma unroll
  for (int j = 0; j < 32; ++j) { mycv[j] = pre[t + j * 256]; myci[j] = t + j * 256; }
  float vmin = mycv[0]; int pmin = 0;
#pragma unroll
  for (int j = 1; j < 32; ++j) { float v = mycv[j]; if (v < vmin) { vmin = v; pmin = j; } }
  for (int j = 32; j < 96; ++j) {
    float v = pre[t + j * 256];
    if (v > vmin) {
      mycv[pmin] = v; myci[pmin] = t + j * 256;
      vmin = mycv[0]; pmin = 0;
#pragma unroll
      for (int q = 1; q < 32; ++q) { float u = mycv[q]; if (u < vmin) { vmin = u; pmin = q; } }
    }
  }

  // cached per-thread argmax
  float lv = mycv[0]; int lgi = myci[0]; int lloc = t * 32;
#pragma unroll
  for (int j = 1; j < 32; ++j) {
    float v = mycv[j]; int g = myci[j];
    if (v > lv || (v == lv && g < lgi)) { lv = v; lgi = g; lloc = t * 32 + j; }
  }
  __syncthreads();

  // phase 2: 32 rounds of block argmax
  for (int r = 0; r < K_TOP; ++r) {
    float v = lv; int gi = lgi; int loc = lloc;
#pragma unroll
    for (int off = 16; off > 0; off >>= 1) {
      float ov = __shfl_xor(v, off);
      int   og = __shfl_xor(gi, off);
      int   ol = __shfl_xor(loc, off);
      if (ov > v || (ov == v && og < gi)) { v = ov; gi = og; loc = ol; }
    }
    if ((t & 31) == 0) { int wd = t >> 5; wvs[wd] = v; wis[wd] = gi; wls[wd] = loc; }
    __syncthreads();
    if (t == 0) {
      float bv = wvs[0]; int bi = wis[0], bl = wls[0];
#pragma unroll
      for (int q = 1; q < 8; ++q)
        if (wvs[q] > bv || (wvs[q] == bv && wis[q] < bi)) { bv = wvs[q]; bi = wis[q]; bl = wls[q]; }
      top_v[r] = bv; top_i[r] = bi; sel = bl;
      cv[bl] = -__builtin_inff();
    }
    __syncthreads();
    if ((sel >> 5) == t) {  // only the owner rescans its list
      lv = mycv[0]; lgi = myci[0]; lloc = t * 32;
#pragma unroll
      for (int j = 1; j < 32; ++j) {
        float vv = mycv[j]; int g = myci[j];
        if (vv > lv || (vv == lv && g < lgi)) { lv = vv; lgi = g; lloc = t * 32 + j; }
      }
    }
    __syncthreads();
  }

  // phase 3: relu, zero row, scatter, emit indices
  if (t < K_TOP) { float bv = top_v[t]; rv[t] = bv > 0.f ? bv : 0.f; }
  __syncthreads();
  float4 z = make_float4(0.f, 0.f, 0.f, 0.f);
  float4* pre4 = reinterpret_cast<float4*>(pre);
  for (int i = t; i < D_SAE / 4; i += 256) pre4[i] = z;
  __syncthreads();
  if (t < K_TOP) {
    pre[top_i[t]] = rv[t];
    idx_out[(size_t)row * K_TOP + t] = (float)top_i[t];
  }

  // phase 4: sparse decode + reconstruction-loss partial
  float lrec = 0.f;
  for (int c = t; c < D_MODEL; c += 256) {
    float acc = b_dec[c];
#pragma unroll
    for (int k = 0; k < K_TOP; ++k)
      acc = fmaf(rv[k], W_dec[(size_t)top_i[k] * D_MODEL + c], acc);
    xrec[(size_t)row * D_MODEL + c] = acc;
    float d = x[(size_t)row * D_MODEL + c] - acc;
    lrec = fmaf(d, d, lrec);
  }
  float lspa = (t < K_TOP) ? rv[t] : 0.f;

#pragma unroll
  for (int off = 16; off > 0; off >>= 1) lrec += __shfl_xor(lrec, off);
  if ((t & 31) == 0) red[t >> 5] = lrec;
  __syncthreads();
  if (t == 0) { float s = 0.f; for (int q = 0; q < 8; ++q) s += red[q]; part_rec[row] = s; }
  __syncthreads();
#pragma unroll
  for (int off = 16; off > 0; off >>= 1) lspa += __shfl_xor(lspa, off);
  if ((t & 31) == 0) red[t >> 5] = lspa;
  __syncthreads();
  if (t == 0) { float s = 0.f; for (int q = 0; q < 8; ++q) s += red[q]; part_spa[row] = s; }
}

// ---------------------------------------------------------------- final loss reduction
__global__ __launch_bounds__(256)
void finalize_losses(const float* __restrict__ part_rec,
                     const float* __restrict__ part_spa,
                     float* __restrict__ out_rec, float* __restrict__ out_spa) {
  __shared__ float red[8];
  const int t = threadIdx.x;
  float s1 = 0.f, s2 = 0.f;
  for (int i = t; i < BATCH; i += 256) { s1 += part_rec[i]; s2 += part_spa[i]; }
#pragma unroll
  for (int off = 16; off > 0; off >>= 1) { s1 += __shfl_xor(s1, off); s2 += __shfl_xor(s2, off); }
  if ((t & 31) == 0) red[t >> 5] = s1;
  __syncthreads();
  if (t == 0) { float s = 0.f; for (int q = 0; q < 8; ++q) s += red[q];
                *out_rec = s / ((float)BATCH * (float)D_MODEL); }
  __syncthreads();
  if ((t & 31) == 0) red[t >> 5] = s2;
  __syncthreads();
  if (t == 0) { float s = 0.f; for (int q = 0; q < 8; ++q) s += red[q];
                *out_spa = s / ((float)BATCH * (float)D_SAE); }
}

// ---------------------------------------------------------------- launch
extern "C" void kernel_launch(void* const* d_in, const int* in_sizes, int n_in,
                              void* d_out, int out_size, void* d_ws, size_t ws_size,
                              hipStream_t stream) {
  const float* x     = (const float*)d_in[0];
  const float* W_enc = (const float*)d_in[1];
  const float* b_enc = (const float*)d_in[2];
  const float* W_dec = (const float*)d_in[3];
  const float* b_dec = (const float*)d_in[4];

  float* out  = (float*)d_out;
  float* xrec = out;                                   // [4096][768]
  float* acts = out + (size_t)BATCH * D_MODEL;         // [4096][24576] (pre_acts, then sparse acts)
  float* idxo = acts + (size_t)BATCH * D_SAE;          // [4096][32] as float
  float* lrec = idxo + (size_t)BATCH * K_TOP;
  float* lspa = lrec + 1;

  unsigned short* Xb = (unsigned short*)d_ws;                       // 4096*768 bf16
  unsigned short* Wb = Xb + (size_t)BATCH * D_MODEL;                // 768*24576 bf16
  float* part_rec = (float*)(Wb + (size_t)D_MODEL * D_SAE);
  float* part_spa = part_rec + BATCH;

  convert_f32_to_bf16<<<1024, 256, 0, stream>>>(x, Xb, BATCH * D_MODEL);
  convert_f32_to_bf16<<<4096, 256, 0, stream>>>(W_enc, Wb, D_MODEL * D_SAE);

  dim3 grid(D_SAE / BN, BATCH / BM);
  encode_gemm_wmma<<<grid, 256, 0, stream>>>(Xb, Wb, b_enc, acts);

  topk_decode<<<BATCH, 256, 0, stream>>>(acts, x, W_dec, b_dec, xrec, idxo,
                                         part_rec, part_spa);
  finalize_losses<<<1, 256, 0, stream>>>(part_rec, part_spa, lrec, lspa);
}
